// ConvolutionalAttention_23888608101032
// MI455X (gfx1250) — matrix operand order repaired and checked by reading, not compiled, not run on hardware
//
#include <hip/hip_runtime.h>
#include <hip/hip_bf16.h>

// ---------------------------------------------------------------------------
// ConvolutionalAttention for MI455X (gfx1250, wave32, WMMA bf16 + TDM)
// ---------------------------------------------------------------------------

typedef __bf16 bf16_t;
typedef bf16_t v16bf __attribute__((ext_vector_type(16)));
typedef bf16_t v8bf  __attribute__((ext_vector_type(8)));
typedef bf16_t v4bf  __attribute__((ext_vector_type(4)));
typedef float  v8f   __attribute__((ext_vector_type(8)));
typedef float  v4f   __attribute__((ext_vector_type(4)));

typedef unsigned int u32x4 __attribute__((ext_vector_type(4)));
typedef int          i32x8 __attribute__((ext_vector_type(8)));
typedef int          i32x4 __attribute__((ext_vector_type(4)));

constexpr int kB = 8;
constexpr int kS = 2048;
constexpr int kD = 1024;
constexpr int kN3 = 3 * kD;        // 3072
constexpr int BK  = 32;            // K-step per WMMA stage
constexpr int LDA = 40;            // padded LDS k-stride in bf16 (80B, 16B aligned)

__device__ __forceinline__ v16bf cat8(v8bf lo, v8bf hi) {
  return __builtin_shufflevector(lo, hi, 0,1,2,3,4,5,6,7,8,9,10,11,12,13,14,15);
}

__device__ __forceinline__ v8f wmma_bf16(v16bf a, v16bf b, v8f c) {
  // D = A(16x32 bf16) * B(32x16 bf16) + C(16x16 f32)
  return __builtin_amdgcn_wmma_f32_16x16x32_bf16(false, a, false, b, (short)0, c,
                                                 false, false);
}

// ---------------------------------------------------------------------------
// Tensor Data Mover: async DMA of a 2D bf16 tile (tile_d0 elems x tile_d1
// rows, row stride stride_elems) from global into LDS at lds_off.
// Optional LDS padding: pad_amount_code DWORD-code inserted every
// pad_interval_code DWORD-interval (ISA 8.4).  Issue from ONE wave, then
// s_wait_tensorcnt 0 before the workgroup barrier.
// This toolchain exposes the 6-arg builtin (g0, g1, g2, g3, g_extra, cpol).
// ---------------------------------------------------------------------------
#define RFL(x) __builtin_amdgcn_readfirstlane((int)(x))
__device__ __forceinline__ void tdm_load_2d_to_lds(
    uint32_t lds_off, const void* gsrc, uint32_t tile_d0, uint32_t tile_d1,
    uint64_t stride_elems, bool pad_en, uint32_t pad_interval_code,
    uint32_t pad_amount_code) {
  uint64_t ga = (uint64_t)(uintptr_t)gsrc;
  uint32_t g0w0 = 1u;                                 // count=1, no gather
  uint32_t g0w1 = lds_off;                            // LDS byte address
  uint32_t g0w2 = (uint32_t)ga;                       // global addr lo
  uint32_t g0w3 = (uint32_t)(ga >> 32) | (2u << 30);  // addr hi | type=2
  uint32_t g1w0 = (1u << 16)                          // data_size = 2 bytes
                | (pad_en ? (1u << 20) : 0u)
                | (pad_interval_code << 22) | (pad_amount_code << 25);
  uint32_t g1w1 = (tile_d0 & 0xFFFFu) << 16;          // tensor_dim0 lo16
  uint32_t g1w2 = ((tile_d0 >> 16) & 0xFFFFu) |       // tensor_dim0 hi16
                  ((tile_d1 & 0xFFFFu) << 16);        // tensor_dim1 lo16
  uint32_t g1w3 = ((tile_d1 >> 16) & 0xFFFFu) |       // tensor_dim1 hi16
                  ((tile_d0 & 0xFFFFu) << 16);        // tile_dim0
  uint32_t g1w4 = (tile_d1 & 0xFFFFu);                // tile_dim1 (dim2=0)
  uint32_t g1w5 = (uint32_t)stride_elems;             // dim0 stride lo32
  uint32_t g1w6 = (uint32_t)((stride_elems >> 32) & 0xFFFFu);
  u32x4 g0 = {(unsigned)RFL(g0w0), (unsigned)RFL(g0w1), (unsigned)RFL(g0w2),
              (unsigned)RFL(g0w3)};
  i32x8 g1 = {RFL(g1w0), RFL(g1w1), RFL(g1w2), RFL(g1w3),
              RFL(g1w4), RFL(g1w5), RFL(g1w6), 0};
  i32x4 z4 = {0, 0, 0, 0};
  i32x8 z8 = {0, 0, 0, 0, 0, 0, 0, 0};
  __builtin_amdgcn_tensor_load_to_lds(g0, g1, z4, z4, z8, 0);
}
#undef RFL

// ---------------------------------------------------------------------------
// Kernel 1: fp32 -> bf16 conversion (vectorized 4-wide)
// ---------------------------------------------------------------------------
__global__ void cvt_f32_bf16(const float* __restrict__ src,
                             bf16_t* __restrict__ dst, int n4) {
  int t = blockIdx.x * blockDim.x + threadIdx.x;
  if (t >= n4) return;
  v4f x = *(const v4f*)(src + (size_t)t * 4);
  v4bf y;
  for (int e = 0; e < 4; e++) y[e] = (bf16_t)x[e];
  *(v4bf*)(dst + (size_t)t * 4) = y;
}

// ---------------------------------------------------------------------------
// Kernel 2: QKV GEMM  qkv[row, 0..3071] = x[row,:] @ W + bias
//   128x128 block tile, BK=32, 8 waves; A-tile staged by TDM (padded),
//   B panel transposed by ALU staging; q scaled 1/sqrt(D) -> bf16, k bf16,
//   v fp32.
// ---------------------------------------------------------------------------
__global__ void __launch_bounds__(256)
qkv_gemm(const bf16_t* __restrict__ X, const bf16_t* __restrict__ W,
         const float* __restrict__ bias,
         bf16_t* __restrict__ Q, bf16_t* __restrict__ K,
         float* __restrict__ V) {
  __shared__ bf16_t Ab[128 * LDA];   // 128 rows x 32 k (TDM-padded stride 40)
  __shared__ bf16_t Bt[128 * LDA];   // transposed: Bt[n][k]
  const int tid = threadIdx.x;
  const int w = tid >> 5, lane = tid & 31;
  const int ln = lane & 15, hi = lane >> 4;
  const int n0 = blockIdx.x * 128;
  const int row0 = blockIdx.y * 128;
  const uint32_t ab_off = (uint32_t)(uintptr_t)&Ab[0];

  v8f acc[8];
  for (int t = 0; t < 8; t++)
    for (int e = 0; e < 8; e++) acc[t][e] = 0.f;

  for (int k0 = 0; k0 < kD; k0 += BK) {
    // A tile 32(k) x 128(rows) via Tensor Data Mover, pad 4 DW / 16 DW -> LDA
    if (w == 0)
      tdm_load_2d_to_lds(ab_off, X + (size_t)row0 * kD + k0, BK, 128,
                         (uint64_t)kD, true, /*interval 16DW*/ 3,
                         /*amount 4DW*/ 3);
    // stage B transposed: read W rows coalesced, write columns
    for (int idx = tid; idx < BK * 128; idx += 256) {
      int k = idx >> 7;
      int n = idx & 127;
      Bt[n * LDA + k] = W[(size_t)(k0 + k) * kN3 + n0 + n];
    }
    if (k0 + BK < kD)   // prefetch next B panel into GL2
      __builtin_prefetch(W + (size_t)(k0 + BK + (tid >> 3)) * kN3 + n0 +
                             (tid & 7) * 16, 0, 1);
    if (w == 0) __builtin_amdgcn_s_wait_tensorcnt((short)0);
    __syncthreads();

    const bf16_t* arow = &Ab[(w * 16 + ln) * LDA];
    const int aoff = hi ? 8 : 0;
    v16bf af = cat8(*(const v8bf*)(arow + aoff),
                    *(const v8bf*)(arow + aoff + 16));
    v16bf bfr[8];
    for (int t = 0; t < 8; t++) {
      const bf16_t* bcol = &Bt[(t * 16 + ln) * LDA + (hi ? 8 : 0)];
      bfr[t] = cat8(*(const v8bf*)(bcol), *(const v8bf*)(bcol + 16));
    }
    for (int t = 0; t < 8; t++) acc[t] = wmma_bf16(af, bfr[t], acc[t]);
    __syncthreads();
  }

  // epilogue: bias add, split into q / k / v
  for (int t = 0; t < 8; t++) {
    int col = n0 + t * 16 + ln;
    float bv = bias[col];
    int seg = col >> 10;       // 0=q 1=k 2=v
    int c = col & (kD - 1);
    for (int e = 0; e < 8; e++) {
      int row = row0 + w * 16 + 8 * hi + e;
      float val = acc[t][e] + bv;
      size_t o = (size_t)row * kD + c;
      if (seg == 0)      Q[o] = (bf16_t)(val * 0.03125f);  // 1/sqrt(1024)
      else if (seg == 1) K[o] = (bf16_t)val;
      else               V[o] = val;
    }
  }
}

// ---------------------------------------------------------------------------
// Kernel 3: conv mix  C = w0*v_prev + w1*v + w2*v_next  (per-channel weights)
//           also emits w2*v_next (bf16) for the diagonal correction
// ---------------------------------------------------------------------------
__global__ void conv_mix(const float* __restrict__ V,
                         const float* __restrict__ Wv,
                         bf16_t* __restrict__ C, bf16_t* __restrict__ W2VN,
                         int n4) {
  int t = blockIdx.x * blockDim.x + threadIdx.x;
  if (t >= n4) return;
  size_t base = (size_t)t * 4;
  int d = (int)(base & (kD - 1));
  size_t row = base / kD;            // flat b*S + s
  int s = (int)(row & (kS - 1));
  v4f v = *(const v4f*)(V + base);
  v4f vp = {0.f, 0.f, 0.f, 0.f};
  v4f vn = {0.f, 0.f, 0.f, 0.f};
  if (s > 0)      vp = *(const v4f*)(V + base - kD);
  if (s < kS - 1) vn = *(const v4f*)(V + base + kD);
  v4bf c, wn;
  for (int e = 0; e < 4; e++) {
    float w0 = Wv[(d + e) * 3 + 0];
    float w1 = Wv[(d + e) * 3 + 1];
    float w2 = Wv[(d + e) * 3 + 2];
    c[e]  = (bf16_t)(w0 * vp[e] + w1 * v[e] + w2 * vn[e]);
    wn[e] = (bf16_t)(w2 * vn[e]);
  }
  *(v4bf*)(C + base) = c;
  *(v4bf*)(W2VN + base) = wn;
}

// ---------------------------------------------------------------------------
// Kernel 4: causal scores pass with TDM double-buffered K tiles.
//   Block = 128 threads (4 waves), 64 query rows per block; each wave owns 16
//   rows and loops only over causal j-tiles of 16 keys.  While the WMMA loop
//   chews tile jt, the Tensor Data Mover streams tile jt+1 into the other
//   LDS buffer (TENSORcnt overlap).  Writes raw fp32 scores + online-softmax
//   stats (m, l) per row.
// ---------------------------------------------------------------------------
__global__ void __launch_bounds__(128)
scores_pass(const bf16_t* __restrict__ Q, const bf16_t* __restrict__ K,
            float* __restrict__ Sb, float* __restrict__ Mb,
            float* __restrict__ Lb) {
  __shared__ bf16_t kbuf[2][16 * kD];   // 2 x 32KB double buffer
  const int tid = threadIdx.x;
  const int w = tid >> 5, lane = tid & 31;
  const int ln = lane & 15, hi = lane >> 4;
  const int b = blockIdx.y;
  const int row0 = blockIdx.x * 64;
  const int wrow0 = row0 + w * 16;
  const bf16_t* Qb = Q + (size_t)b * kS * kD;
  const bf16_t* Kb = K + (size_t)b * kS * kD;
  float* Srow = Sb + (size_t)b * kS * kS;

  float mrow[8], lrow[8];
  for (int r = 0; r < 8; r++) { mrow[r] = -3.0e38f; lrow[r] = 0.f; }

  const int njt = row0 / 16 + 4;     // causal limit for this block
  const bf16_t* qrow = Qb + (size_t)(wrow0 + ln) * kD;

  // prologue: DMA tile 0 into buffer 0
  if (w == 0) {
    tdm_load_2d_to_lds((uint32_t)(uintptr_t)&kbuf[0][0], Kb, kD, 16,
                       (uint64_t)kD, false, 0, 0);
    __builtin_amdgcn_s_wait_tensorcnt((short)0);
  }
  __syncthreads();

  for (int jt = 0; jt < njt; jt++) {
    const int cur = jt & 1;
    // kick off DMA of the next K tile into the other buffer
    if (w == 0 && jt + 1 < njt)
      tdm_load_2d_to_lds((uint32_t)(uintptr_t)&kbuf[cur ^ 1][0],
                         Kb + (size_t)(jt + 1) * 16 * kD, kD, 16,
                         (uint64_t)kD, false, 0, 0);

    v8f c;
    for (int e = 0; e < 8; e++) c[e] = 0.f;
    for (int kk = 0; kk < kD; kk += 32) {
      int aoff = kk + (hi ? 8 : 0);
      v16bf af = cat8(*(const v8bf*)(qrow + aoff),
                      *(const v8bf*)(qrow + aoff + 16));
      const bf16_t* kcol = &kbuf[cur][ln * kD + kk + (hi ? 8 : 0)];
      v16bf bfr = cat8(*(const v8bf*)(kcol), *(const v8bf*)(kcol + 16));
      c = wmma_bf16(af, bfr, c);
    }

    // causal mask + raw score store + online softmax stats
    int jg = jt * 16 + ln;
    for (int r = 0; r < 8; r++) {
      int i = wrow0 + r + 8 * hi;
      float s = c[r];
      if (jg <= i) Srow[(size_t)i * kS + jg] = s;
      else         s = -3.0e38f;
      float tmax = s;                      // 16-lane butterfly max
      tmax = fmaxf(tmax, __shfl_xor(tmax, 1));
      tmax = fmaxf(tmax, __shfl_xor(tmax, 2));
      tmax = fmaxf(tmax, __shfl_xor(tmax, 4));
      tmax = fmaxf(tmax, __shfl_xor(tmax, 8));
      float newm = fmaxf(mrow[r], tmax);
      float p = __expf(s - newm);
      p += __shfl_xor(p, 1);               // 16-lane butterfly sum
      p += __shfl_xor(p, 2);
      p += __shfl_xor(p, 4);
      p += __shfl_xor(p, 8);
      lrow[r] = lrow[r] * __expf(mrow[r] - newm) + p;
      mrow[r] = newm;
    }

    if (w == 0) __builtin_amdgcn_s_wait_tensorcnt((short)0);
    __syncthreads();
  }

  if (ln == 0) {
    for (int r = 0; r < 8; r++) {
      int i = wrow0 + r + 8 * hi;
      Mb[(size_t)b * kS + i] = mrow[r];
      Lb[(size_t)b * kS + i] = lrow[r];
    }
  }
}

// ---------------------------------------------------------------------------
// Kernel 5: softmax normalize  A = exp(s - m) / l  (bf16), zero upper tri,
//           capture diag(A).
// ---------------------------------------------------------------------------
__global__ void softmax_norm(const float* __restrict__ Sb,
                             const float* __restrict__ Mb,
                             const float* __restrict__ Lb,
                             bf16_t* __restrict__ A,
                             float* __restrict__ DiagA) {
  int fid = blockIdx.x;                    // flat b*S + i
  int i = fid & (kS - 1);
  int j0 = blockIdx.y * 1024 + threadIdx.x * 4;
  float m = Mb[fid];
  float linv = 1.0f / Lb[fid];
  const float* srow = Sb + (size_t)fid * kS;
  bf16_t* arow = A + (size_t)fid * kS;
  v4f s = {0.f, 0.f, 0.f, 0.f};
  if (j0 <= i) s = *(const v4f*)(srow + j0);
  v4bf out;
  for (int e = 0; e < 4; e++) {
    int j = j0 + e;
    float a = 0.f;
    if (j <= i) {
      a = __expf(s[e] - m) * linv;
      if (j == i) DiagA[fid] = a;
    }
    out[e] = (bf16_t)a;
  }
  *(v4bf*)(arow + j0) = out;
}

// ---------------------------------------------------------------------------
// Kernel 6: out = A @ C  with causal K-loop truncation and fused epilogue
//           out[i,d] -= diagA[i] * (w2*v_next)[i,d];  fp32 output.
//           A tile staged by TDM (padded), C panel transposed by ALU staging.
// ---------------------------------------------------------------------------
__global__ void __launch_bounds__(256)
av_gemm(const bf16_t* __restrict__ A, const bf16_t* __restrict__ C,
        const float* __restrict__ DiagA, const bf16_t* __restrict__ W2VN,
        float* __restrict__ Out) {
  __shared__ bf16_t Ab[128 * LDA];
  __shared__ bf16_t Bt[128 * LDA];
  const int tid = threadIdx.x;
  const int w = tid >> 5, lane = tid & 31;
  const int ln = lane & 15, hi = lane >> 4;
  const int b = blockIdx.z;
  const int n0 = blockIdx.x * 128;
  const int row0 = blockIdx.y * 128;
  const bf16_t* Abase = A + (size_t)b * kS * kS;
  const bf16_t* Cbase = C + (size_t)b * kS * kD;
  const uint32_t ab_off = (uint32_t)(uintptr_t)&Ab[0];

  v8f acc[8];
  for (int t = 0; t < 8; t++)
    for (int e = 0; e < 8; e++) acc[t][e] = 0.f;

  const int kend = row0 + 128;       // A[i,j]==0 for j>i -> skip upper tiles
  for (int k0 = 0; k0 < kend; k0 += BK) {
    if (w == 0)
      tdm_load_2d_to_lds(ab_off, Abase + (size_t)row0 * kS + k0, BK, 128,
                         (uint64_t)kS, true, 3, 3);
    for (int idx = tid; idx < BK * 128; idx += 256) {
      int k = idx >> 7;
      int n = idx & 127;
      Bt[n * LDA + k] = Cbase[(size_t)(k0 + k) * kD + n0 + n];
    }
    if (k0 + BK < kend)  // prefetch next C panel into GL2
      __builtin_prefetch(Cbase + (size_t)(k0 + BK + (tid >> 3)) * kD + n0 +
                             (tid & 7) * 16, 0, 1);
    if (w == 0) __builtin_amdgcn_s_wait_tensorcnt((short)0);
    __syncthreads();

    const bf16_t* arow = &Ab[(w * 16 + ln) * LDA];
    const int aoff = hi ? 8 : 0;
    v16bf af = cat8(*(const v8bf*)(arow + aoff),
                    *(const v8bf*)(arow + aoff + 16));
    v16bf bfr[8];
    for (int t = 0; t < 8; t++) {
      const bf16_t* bcol = &Bt[(t * 16 + ln) * LDA + (hi ? 8 : 0)];
      bfr[t] = cat8(*(const v8bf*)(bcol), *(const v8bf*)(bcol + 16));
    }
    for (int t = 0; t < 8; t++) acc[t] = wmma_bf16(af, bfr[t], acc[t]);
    __syncthreads();
  }

  for (int e = 0; e < 8; e++) {
    int row = row0 + w * 16 + 8 * hi + e;
    float dg = DiagA[(size_t)b * kS + row];
    for (int t = 0; t < 8; t++) {
      int col = n0 + t * 16 + ln;
      size_t o = ((size_t)b * kS + row) * kD + col;
      Out[o] = acc[t][e] - dg * (float)W2VN[o];
    }
  }
}

// ---------------------------------------------------------------------------
// launch
// ---------------------------------------------------------------------------
extern "C" void kernel_launch(void* const* d_in, const int* in_sizes, int n_in,
                              void* d_out, int out_size, void* d_ws,
                              size_t ws_size, hipStream_t stream) {
  (void)in_sizes; (void)n_in; (void)out_size; (void)ws_size;
  const float* x    = (const float*)d_in[0];   // (B,S,D)
  const float* Wq   = (const float*)d_in[1];   // (D,3D)
  const float* bq   = (const float*)d_in[2];   // (3D,)
  const float* Wv   = (const float*)d_in[3];   // (D,3)

  char* ws = (char*)d_ws;
  size_t off = 0;
  auto alloc = [&](size_t bytes) -> void* {
    void* p = ws + off;
    off += (bytes + 255) & ~(size_t)255;
    return p;
  };
  const size_t nBSD = (size_t)kB * kS * kD;    // 16,777,216
  const size_t nW   = (size_t)kD * kN3;        //  3,145,728
  const size_t nBSS = (size_t)kB * kS * kS;    // 33,554,432
  const size_t nBS  = (size_t)kB * kS;         //      16,384

  bf16_t* xb   = (bf16_t*)alloc(nBSD * 2);
  bf16_t* wb   = (bf16_t*)alloc(nW * 2);
  bf16_t* qb   = (bf16_t*)alloc(nBSD * 2);
  bf16_t* kb   = (bf16_t*)alloc(nBSD * 2);
  float*  vb   = (float*) alloc(nBSD * 4);
  bf16_t* cb   = (bf16_t*)alloc(nBSD * 2);
  bf16_t* w2vn = (bf16_t*)alloc(nBSD * 2);
  float*  sb   = (float*) alloc(nBSS * 4);     // raw scores (L2-resident)
  bf16_t* ab   = (bf16_t*)alloc(nBSS * 2);     // normalized A
  float*  mb   = (float*) alloc(nBS * 4);
  float*  lb   = (float*) alloc(nBS * 4);
  float*  dgb  = (float*) alloc(nBS * 4);

  // 1) convert inputs to bf16
  cvt_f32_bf16<<<(int)(nBSD / 4 / 256), 256, 0, stream>>>(x, xb, (int)(nBSD / 4));
  cvt_f32_bf16<<<(int)(nW / 4 / 256), 256, 0, stream>>>(Wq, wb, (int)(nW / 4));

  // 2) qkv = x @ W + b   (q pre-scaled by 1/sqrt(D))
  qkv_gemm<<<dim3(kN3 / 128, (kB * kS) / 128), 256, 0, stream>>>(
      xb, wb, bq, qb, kb, vb);

  // 3) conv mix
  conv_mix<<<(int)(nBSD / 4 / 256), 256, 0, stream>>>(vb, Wv, cb, w2vn,
                                                      (int)(nBSD / 4));

  // 4) causal scores + online softmax stats (TDM double-buffered)
  scores_pass<<<dim3(kS / 64, kB), 128, 0, stream>>>(qb, kb, sb, mb, lb);

  // 5) normalize to A (bf16), capture diag
  softmax_norm<<<dim3(kB * kS, kS / 1024), 256, 0, stream>>>(sb, mb, lb, ab,
                                                             dgb);

  // 6) out = A @ C - diag(A) * (w2*v_next)
  av_gemm<<<dim3(kD / 128, kS / 128, kB), 256, 0, stream>>>(
      ab, cb, dgb, w2vn, (float*)d_out);
}
